// SparsePPGN_4380866642582
// MI455X (gfx1250) — compile-verified, matrix operand order checked by reading, not run on hardware
//
#include <hip/hip_runtime.h>
#include <hip/hip_bf16.h>
#include <math.h>

// ---------------------------------------------------------------------------
// SparsePPGN forward for gfx1250 (MI455X).
// GEMMs via v_wmma_f32_16x16x32_f16 (f16 in, f32 accum), 32x64 tile per wave,
// B-operand double-buffered in LDS (shared by 8 waves, 1 barrier/k-step),
// A prefetched with global_prefetch; triangle messages via global_atomic_add_f32.
//
// Assumed d_in layout (setup_inputs() insertion order, params flattened in
// nested insertion order, "W" before "b"):
//   0 edge_index[2E] 1 batch[n] 2 triangle_index[3T] 3 rev_index[E]
//   4 node_attr[n*8] 5 edge_attr[E*4] 6 node_emb[n*64]
//   7 noise_cond[B] 8 red_frac[B] 9 target_size[B]
//  10/11 node_emb.W/b  12/13 edge_emb.W/b  14/15 noise_emb.W/b 16/17 red_emb.W/b
//  18..21 node_in_mlp (W0,b0,W1,b1)   22..25 edge_in_mlp
//  26+12*l .. : layer l: mlp1 W0,b0,W1,b1 ; mlp2 W0,b0,W1,b1 ; mlp3 W0,b0,W1,b1
//  62/63 node_out.W/b  64/65 edge_out.W/b
// ---------------------------------------------------------------------------

typedef __attribute__((ext_vector_type(16))) _Float16 v16h;
typedef __attribute__((ext_vector_type(8)))  _Float16 v8h;
typedef __attribute__((ext_vector_type(4)))  _Float16 v4h;
typedef __attribute__((ext_vector_type(8)))  float    v8f;

#define BSTRIDE 40            // halves per B-row in LDS (80 B: 16-lane stride -> 16 banks)
#define BTILE   (64 * BSTRIDE)

// Assemble one 16x32 fragment (CDNA5 16-bit A/B layout) from two 16 B chunks:
// lane holds K in {hs..hs+7} U {hs+16..hs+23}, p already offset by r*ld + hs.
__device__ __forceinline__ v16h frag_pair(const _Float16* __restrict__ p) {
  v8h lo = *(const v8h*)(p);
  v8h hi = *(const v8h*)(p + 16);
  v16h f;
#pragma unroll
  for (int i = 0; i < 8; ++i) { f[i] = lo[i]; f[8 + i] = hi[i]; }
  return f;
}

__device__ __forceinline__ void store_tile(v8f acc, _Float16* __restrict__ C,
                                           const float* __restrict__ bias,
                                           int rowb, int col, int N,
                                           int Mreal, int relu) {
  float bv = bias[col];
#pragma unroll
  for (int r = 0; r < 8; ++r) {
    int row = rowb + r;
    if (row < Mreal) {
      float v = acc[r] + bv;
      if (relu && v < 0.f) v = 0.f;
      C[(size_t)row * N + col] = (_Float16)v;
    }
  }
}

// C[Mpad,N] = relu?(A[Mpad,K] @ Wt^T + bias); Wt is f16 [N,K] (pre-transposed).
// Block: 8 waves; each wave computes a 32x64 tile (8 WMMA accumulators).
// The 64x32 B-tile for the block's columns is double-buffered in LDS.
__global__ __launch_bounds__(256)
void wmma_gemm_kernel(const _Float16* __restrict__ A,
                      const _Float16* __restrict__ Wt,
                      const float* __restrict__ bias,
                      _Float16* __restrict__ C,
                      int Mreal, int K, int N, int relu) {
  __shared__ _Float16 Bs[2 * BTILE];
  const int lane = threadIdx.x & 31;
  const int wave = threadIdx.x >> 5;
  const int row0 = (blockIdx.x * 8 + wave) * 32;   // grid sized exactly: row0 < Mpad
  const int col0 = (int)blockIdx.y << 6;

  // cooperative B staging: 256 threads x 16 B = 64 rows x 32 halves
  const int sn = threadIdx.x >> 2;                 // 0..63  (B row = col0+sn)
  const int sp = (threadIdx.x & 3) << 3;           // 0,8,16,24 (k-halves)
  const _Float16* bsrc = Wt + (size_t)(col0 + sn) * K + sp;
  _Float16* bdst = Bs + sn * BSTRIDE + sp;

  const int r  = lane & 15;
  const int hs = (lane >> 4) << 3;
  const _Float16* a0p = A + (size_t)(row0 + r) * K + hs;
  const _Float16* a1p = A + (size_t)(row0 + 16 + r) * K + hs;
  const _Float16* bfp = Bs + r * BSTRIDE + hs;

  v8f acc[8];
#pragma unroll
  for (int i = 0; i < 8; ++i) acc[i] = (v8f){};

  // prologue: stage first B tile into buffer 0
  *(v8h*)bdst = *(const v8h*)bsrc;
  __syncthreads();

  for (int k0 = 0; k0 < K; k0 += 32) {
    const int cur = (k0 >> 5) & 1;
    if (k0 + 32 < K) {                             // stage next tile into other buffer
      *(v8h*)(bdst + (cur ^ 1) * BTILE) = *(const v8h*)(bsrc + k0 + 32);
      __builtin_prefetch(a0p + k0 + 32, 0, 3);     // prefetch next A rows
      __builtin_prefetch(a1p + k0 + 32, 0, 3);
    }

    v16h a0 = frag_pair(a0p + k0);
    v16h a1 = frag_pair(a1p + k0);
    const _Float16* bb = bfp + cur * BTILE;
    v16h b0 = frag_pair(bb);
    v16h b1 = frag_pair(bb + 16 * BSTRIDE);
    v16h b2 = frag_pair(bb + 32 * BSTRIDE);
    v16h b3 = frag_pair(bb + 48 * BSTRIDE);

    acc[0] = __builtin_amdgcn_wmma_f32_16x16x32_f16(false, a0, false, b0, (short)0, acc[0], false, false);
    acc[4] = __builtin_amdgcn_wmma_f32_16x16x32_f16(false, a1, false, b0, (short)0, acc[4], false, false);
    acc[1] = __builtin_amdgcn_wmma_f32_16x16x32_f16(false, a0, false, b1, (short)0, acc[1], false, false);
    acc[5] = __builtin_amdgcn_wmma_f32_16x16x32_f16(false, a1, false, b1, (short)0, acc[5], false, false);
    acc[2] = __builtin_amdgcn_wmma_f32_16x16x32_f16(false, a0, false, b2, (short)0, acc[2], false, false);
    acc[6] = __builtin_amdgcn_wmma_f32_16x16x32_f16(false, a1, false, b2, (short)0, acc[6], false, false);
    acc[3] = __builtin_amdgcn_wmma_f32_16x16x32_f16(false, a0, false, b3, (short)0, acc[3], false, false);
    acc[7] = __builtin_amdgcn_wmma_f32_16x16x32_f16(false, a1, false, b3, (short)0, acc[7], false, false);

    // publishes next buffer + retires this buffer's reads before overwrite
    __syncthreads();
  }

  const int ncol = lane & 15;
  const int mofs = (lane >> 4) << 3;
#pragma unroll
  for (int t = 0; t < 4; ++t) {
    store_tile(acc[t],     C, bias, row0 + mofs,      col0 + t * 16 + ncol, N, Mreal, relu);
    store_tile(acc[4 + t], C, bias, row0 + 16 + mofs, col0 + t * 16 + ncol, N, Mreal, relu);
  }
}

// W[K,N] f32 -> Wt[N,K] f16 (transposed for the WMMA B operand).
__global__ void wconv_kernel(const float* __restrict__ W,
                             _Float16* __restrict__ Wt, int K, int N) {
  int idx = blockIdx.x * 256 + threadIdx.x;
  if (idx >= K * N) return;
  int k = idx / N, ncol = idx % N;
  Wt[(size_t)ncol * K + k] = (_Float16)W[idx];
}

__global__ void zero_f32_kernel(float* p, long cnt) {
  long i = (long)blockIdx.x * 256 + threadIdx.x;
  if (i < cnt) p[i] = 0.f;
}
__global__ void zero_i32_kernel(int* p, long cnt) {
  long i = (long)blockIdx.x * 256 + threadIdx.x;
  if (i < cnt) p[i] = 0;
}

// Per-graph conditioning features: [B,192] = [noise_emb | red_emb | pos_enc].
__global__ void cond_kernel(const float* noise, const float* Wn, const float* bn,
                            const float* redf, const float* Wr, const float* br,
                            const float* tgt, float* cond, int B) {
  int i = blockIdx.x * 256 + threadIdx.x;
  if (i >= B * 192) return;
  int b = i / 192, c = i % 192;
  float v;
  if (c < 64)       v = noise[b] * Wn[c] + bn[c];
  else if (c < 128) v = redf[b] * Wr[c - 64] + br[c - 64];
  else {
    int cc = c - 128, h = cc & 31;
    float freq = expf(-logf(10000.f) * (float)h / 32.f);
    float ang = tgt[b] * freq;
    v = (cc < 32) ? sinf(ang) : cosf(ang);
  }
  cond[i] = v;
}

// x_node input features f16 [n,320] = [na | node_emb | cond[batch]].
__global__ void node_in_kernel(const float* __restrict__ node_attr,
                               const float* __restrict__ Wne, const float* __restrict__ bne,
                               const float* __restrict__ node_emb,
                               const int* __restrict__ batch,
                               const float* __restrict__ cond,
                               _Float16* __restrict__ out, int n) {
  int idx = blockIdx.x * 256 + threadIdx.x;
  if (idx >= n * 320) return;
  int i = idx / 320, c = idx % 320;
  float v;
  if (c < 64) {
    v = bne[c];
    const float* at = node_attr + (size_t)i * 8;
#pragma unroll
    for (int k = 0; k < 8; ++k) v += at[k] * Wne[k * 64 + c];
  } else if (c < 128) v = node_emb[(size_t)i * 64 + (c - 64)];
  else                v = cond[batch[i] * 192 + (c - 128)];
  out[idx] = (_Float16)v;
}

// x_edge input features f16 [E,384] = [ea | emb[src] | emb[dst] | cond[batch[src]]].
__global__ void edge_in_kernel(const float* __restrict__ edge_attr,
                               const float* __restrict__ Wee, const float* __restrict__ bee,
                               const float* __restrict__ node_emb,
                               const int* __restrict__ edge_index, int E,
                               const int* __restrict__ batch,
                               const float* __restrict__ cond,
                               _Float16* __restrict__ out) {
  int idx = blockIdx.x * 256 + threadIdx.x;
  if (idx >= E * 384) return;
  int e = idx / 384, c = idx % 384;
  int s = edge_index[e], d = edge_index[E + e];
  float v;
  if (c < 64) {
    v = bee[c];
    const float* at = edge_attr + (size_t)e * 4;
#pragma unroll
    for (int k = 0; k < 4; ++k) v += at[k] * Wee[k * 64 + c];
  } else if (c < 192) {
    int node = (c < 128) ? s : d;
    v = node_emb[(size_t)node * 64 + ((c - 64) & 63)];
  } else v = cond[batch[s] * 192 + (c - 192)];
  out[idx] = (_Float16)v;
}

__global__ void count_kernel(const int* __restrict__ t2, int* __restrict__ counts, int T) {
  int i = blockIdx.x * 256 + threadIdx.x;
  if (i < T) atomicAdd(&counts[t2[i]], 1);
}
__global__ void norm_kernel(const int* __restrict__ counts, float* __restrict__ nrm, int Mp) {
  int i = blockIdx.x * 256 + threadIdx.x;
  if (i < Mp) nrm[i] = rsqrtf((float)(counts[i] > 0 ? counts[i] : 1));
}

// msg[t2] += m1[t0] * m2[t1]   (one wave per triple, 4 channels per lane)
__global__ __launch_bounds__(256)
void tri_kernel(const int* __restrict__ t0, const int* __restrict__ t1,
                const int* __restrict__ t2,
                const _Float16* __restrict__ m1, const _Float16* __restrict__ m2,
                float* __restrict__ msg, int T) {
  int tri = blockIdx.x * 8 + (threadIdx.x >> 5);
  if (tri >= T) return;
  int lane = threadIdx.x & 31;
  int a = t0[tri], b = t1[tri], c = t2[tri];
  int ch = lane * 4;
  v4h x = *(const v4h*)(m1 + (size_t)a * 128 + ch);
  v4h y = *(const v4h*)(m2 + (size_t)b * 128 + ch);
  float* dst = msg + (size_t)c * 128 + ch;
#pragma unroll
  for (int i = 0; i < 4; ++i) atomicAdd(dst + i, (float)x[i] * (float)y[i]);
}

// mlp3 input f16 [Mp,384] = [ x | msg * rsqrt(count) ]
__global__ void concat_kernel(const _Float16* __restrict__ X,
                              const float* __restrict__ msg,
                              const float* __restrict__ nrm,
                              _Float16* __restrict__ out, long total) {
  long idx = (long)blockIdx.x * 256 + threadIdx.x;
  if (idx >= total) return;
  int row = (int)(idx / 384), c = (int)(idx % 384);
  _Float16 v;
  if (c < 256) v = X[(size_t)row * 256 + c];
  else         v = (_Float16)(msg[(size_t)row * 128 + (c - 256)] * nrm[row]);
  out[idx] = v;
}

// out[rows,O] = concat(X0..X3)[rows,1024] @ W[1024,O] + b   (small-N VALU GEMM)
__global__ void out_lin_kernel(const _Float16* __restrict__ X0, const _Float16* __restrict__ X1,
                               const _Float16* __restrict__ X2, const _Float16* __restrict__ X3,
                               const float* __restrict__ W, const float* __restrict__ bias,
                               float* __restrict__ out, int rows, int O) {
  int idx = blockIdx.x * 256 + threadIdx.x;
  if (idx >= rows * O) return;
  int i = idx / O, o = idx % O;
  const _Float16* xs[4] = {X0 + (size_t)i * 256, X1 + (size_t)i * 256,
                           X2 + (size_t)i * 256, X3 + (size_t)i * 256};
  float acc = bias[o];
#pragma unroll 1
  for (int s = 0; s < 4; ++s) {
    const _Float16* xr = xs[s];
    const float* wr = W + (size_t)(s * 256) * O + o;
    for (int c = 0; c < 256; ++c) acc += (float)xr[c] * wr[(size_t)c * O];
  }
  out[idx] = acc;
}

// out_edge = 0.5 * (oe + oe[rev_index])
__global__ void edge_combine_kernel(const float* __restrict__ oe,
                                    const int* __restrict__ rev,
                                    float* __restrict__ out, int E) {
  int idx = blockIdx.x * 256 + threadIdx.x;
  if (idx >= E * 4) return;
  int e = idx / 4, o = idx % 4;
  out[idx] = 0.5f * (oe[idx] + oe[(size_t)rev[e] * 4 + o]);
}

// ---------------------------------------------------------------------------

extern "C" void kernel_launch(void* const* d_in, const int* in_sizes, int n_in,
                              void* d_out, int out_size, void* d_ws, size_t ws_size,
                              hipStream_t stream) {
  (void)n_in; (void)out_size; (void)ws_size;
  const int n = in_sizes[1];
  const int E = in_sizes[3];
  const int T = in_sizes[2] / 3;
  const int B = in_sizes[7];
  const int M = n + E;
  auto pad = [](int x) { return (x + 255) & ~255; };   // 8 waves x 32 rows
  const int npad = pad(n), Epad = pad(E), Mpad = pad(M);

  const int*   edge_index = (const int*)d_in[0];
  const int*   batch      = (const int*)d_in[1];
  const int*   tri        = (const int*)d_in[2];
  const int*   rev        = (const int*)d_in[3];
  const float* node_attr  = (const float*)d_in[4];
  const float* edge_attr  = (const float*)d_in[5];
  const float* node_embed = (const float*)d_in[6];
  const float* noise      = (const float*)d_in[7];
  const float* redf       = (const float*)d_in[8];
  const float* tgt        = (const float*)d_in[9];
  auto P = [&](int i) { return (const float*)d_in[i]; };

  // ----- workspace carve-up (requires ~700 MB; 432 GiB device) -----
  size_t off = 0;
  auto alloc = [&](size_t bytes) -> void* {
    void* p = (char*)d_ws + off;
    off += (bytes + 255) & ~(size_t)255;
    return p;
  };
  struct WtMat { _Float16* p; int K, N; };
  auto allocWt = [&](int K, int N) { WtMat m; m.p = (_Float16*)alloc((size_t)K * N * 2); m.K = K; m.N = N; return m; };

  float* cond = (float*)alloc((size_t)B * 192 * 4);
  WtMat wt_ni0 = allocWt(320, 256), wt_ni1 = allocWt(256, 256);
  WtMat wt_ei0 = allocWt(384, 256), wt_ei1 = allocWt(256, 256);
  WtMat wt_l[3][6];
  for (int l = 0; l < 3; ++l) {
    wt_l[l][0] = allocWt(256, 256); wt_l[l][1] = allocWt(256, 128);
    wt_l[l][2] = allocWt(256, 256); wt_l[l][3] = allocWt(256, 128);
    wt_l[l][4] = allocWt(384, 256); wt_l[l][5] = allocWt(256, 256);
  }
  _Float16* Xin_node = (_Float16*)alloc((size_t)npad * 320 * 2);
  _Float16* Xin_edge = (_Float16*)alloc((size_t)Epad * 384 * 2);
  _Float16* Hidden   = (_Float16*)alloc((size_t)Mpad * 256 * 2);
  _Float16* X[4];
  for (int s = 0; s < 4; ++s) X[s] = (_Float16*)alloc((size_t)Mpad * 256 * 2);
  _Float16* m1  = (_Float16*)alloc((size_t)Mpad * 128 * 2);
  _Float16* m2  = (_Float16*)alloc((size_t)Mpad * 128 * 2);
  float*    msg = (float*)alloc((size_t)Mpad * 128 * 4);
  _Float16* in384 = (_Float16*)alloc((size_t)Mpad * 384 * 2);
  int*      counts = (int*)alloc((size_t)Mpad * 4);
  float*    normv  = (float*)alloc((size_t)Mpad * 4);
  float*    oe     = (float*)alloc((size_t)Epad * 4 * 4);

  // ----- 1. weight transpose+convert to f16 -----
  auto wconv = [&](int srcIdx, WtMat& m) {
    int total = m.K * m.N;
    wconv_kernel<<<(total + 255) / 256, 256, 0, stream>>>(P(srcIdx), m.p, m.K, m.N);
  };
  wconv(18, wt_ni0); wconv(20, wt_ni1);
  wconv(22, wt_ei0); wconv(24, wt_ei1);
  for (int l = 0; l < 3; ++l) {
    int base = 26 + 12 * l;
    wconv(base + 0, wt_l[l][0]); wconv(base + 2,  wt_l[l][1]);
    wconv(base + 4, wt_l[l][2]); wconv(base + 6,  wt_l[l][3]);
    wconv(base + 8, wt_l[l][4]); wconv(base + 10, wt_l[l][5]);
  }

  // ----- 2. conditioning + input features -----
  cond_kernel<<<(B * 192 + 255) / 256, 256, 0, stream>>>(
      noise, P(14), P(15), redf, P(16), P(17), tgt, cond, B);
  node_in_kernel<<<(n * 320 + 255) / 256, 256, 0, stream>>>(
      node_attr, P(10), P(11), node_embed, batch, cond, Xin_node, n);
  edge_in_kernel<<<(E * 384 + 255) / 256, 256, 0, stream>>>(
      edge_attr, P(12), P(13), node_embed, edge_index, E, batch, cond, Xin_edge);

  // ----- 3. message counts -> norm (rsqrt) -----
  zero_i32_kernel<<<(Mpad + 255) / 256, 256, 0, stream>>>(counts, Mpad);
  count_kernel<<<(T + 255) / 256, 256, 0, stream>>>(tri + 2 * (size_t)T, counts, T);
  norm_kernel<<<(Mpad + 255) / 256, 256, 0, stream>>>(counts, normv, Mpad);

  // ----- GEMM launcher: block covers 256 rows x 64 cols -----
  auto gemm = [&](const _Float16* A, const WtMat& w, const float* bias,
                  _Float16* C, int Mp, int Mreal, int relu) {
    dim3 grid(Mp / 256, w.N / 64);
    wmma_gemm_kernel<<<grid, 256, 0, stream>>>(A, w.p, bias, C, Mreal, w.K, w.N, relu);
  };

  // ----- 4. input MLPs -> X0 = [x_node; x_edge] -----
  gemm(Xin_node, wt_ni0, P(19), Hidden, npad, n, 1);
  gemm(Hidden,   wt_ni1, P(21), X[0],   npad, n, 0);
  gemm(Xin_edge, wt_ei0, P(23), Hidden, Epad, E, 1);
  gemm(Hidden,   wt_ei1, P(25), X[0] + (size_t)n * 256, Epad, E, 0);

  // ----- 5. PPGN layers -----
  const long msg_cnt = (long)Mpad * 128;
  const long cat_cnt = (long)Mpad * 384;
  for (int l = 0; l < 3; ++l) {
    int base = 26 + 12 * l;
    gemm(X[l],   wt_l[l][0], P(base + 1), Hidden, Mpad, M, 1);
    gemm(Hidden, wt_l[l][1], P(base + 3), m1,     Mpad, M, 0);
    gemm(X[l],   wt_l[l][2], P(base + 5), Hidden, Mpad, M, 1);
    gemm(Hidden, wt_l[l][3], P(base + 7), m2,     Mpad, M, 0);

    zero_f32_kernel<<<(unsigned)((msg_cnt + 255) / 256), 256, 0, stream>>>(msg, msg_cnt);
    tri_kernel<<<(T + 7) / 8, 256, 0, stream>>>(
        tri, tri + (size_t)T, tri + 2 * (size_t)T, m1, m2, msg, T);
    concat_kernel<<<(unsigned)((cat_cnt + 255) / 256), 256, 0, stream>>>(
        X[l], msg, normv, in384, cat_cnt);

    gemm(in384,  wt_l[l][4], P(base + 9),  Hidden,   Mpad, M, 1);
    gemm(Hidden, wt_l[l][5], P(base + 11), X[l + 1], Mpad, M, 0);
  }

  // ----- 6. output heads -----
  float* out_node = (float*)d_out;
  float* out_edge = (float*)d_out + (size_t)n * 8;
  out_lin_kernel<<<(n * 8 + 255) / 256, 256, 0, stream>>>(
      X[0], X[1], X[2], X[3], P(62), P(63), out_node, n, 8);
  const size_t eoff = (size_t)n * 256;
  out_lin_kernel<<<(E * 4 + 255) / 256, 256, 0, stream>>>(
      X[0] + eoff, X[1] + eoff, X[2] + eoff, X[3] + eoff, P(64), P(65), oe, E, 4);
  edge_combine_kernel<<<(E * 4 + 255) / 256, 256, 0, stream>>>(oe, rev, out_edge, E);
}
